// SlotAttention_91268055040004
// MI455X (gfx1250) — compile-verified, matrix operand order checked by reading, not run on hardware
//
#include <hip/hip_runtime.h>

// ---------------------------------------------------------------------------
// Slot Attention for MI455X (gfx1250, wave32, WMMA bf16 16x16x32)
// b=128, n=1024, d=512, S=100, 3 iterations.
// All GEMMs are NT form: C[M,N] = A[M,K] (row-major) * W[N,K]^T (row-major),
// mapped onto v_wmma_f32_16x16x32_bf16. Wave tile = 32x64 (8 WMMAs/k-step).
// ---------------------------------------------------------------------------

typedef __bf16 bf16;
typedef __attribute__((ext_vector_type(16))) __bf16 v16bf;
typedef __attribute__((ext_vector_type(8)))  __bf16 v8bf;
typedef __attribute__((ext_vector_type(8)))  float  v8f;

#define B_   128
#define N_   1024
#define D_   512
#define S_   100
#define TD_  1536          // 3*D
#define BS_  (B_ * S_)     // 12800
#define SCALE_ 0.044194173824159216f   // 512^-0.5

__device__ __forceinline__ bf16 to_bf16(float f) {
    unsigned u = __builtin_bit_cast(unsigned, f);
    unsigned r = (u + 0x7FFFu + ((u >> 16) & 1u)) >> 16;  // round-nearest-even
    unsigned short h = (unsigned short)r;
    return __builtin_bit_cast(bf16, h);
}

// A/B fragment for 16-bit WMMA (ISA 7.12.2):
// lanes 0-15 : row/col = lane,     K = kb+0..7, kb+16..23
// lanes 16-31: row/col = lane-16,  K = kb+8..15, kb+24..31
// p already points at (row, k0 + half*8): two contiguous 16B chunks.
__device__ __forceinline__ v16bf ldfrag(const bf16* p) {
    union { v16bf v; v8bf h[2]; } u;
    u.h[0] = *reinterpret_cast<const v8bf*>(p);
    u.h[1] = *reinterpret_cast<const v8bf*>(p + 16);
    return u.v;
}

// ---------------------------------------------------------------------------
// NT GEMM: block = 4 waves, block tile 128x64. Each wave owns a 32x64 strip:
// 2 A fragments x 4 B fragments -> 8 WMMAs per 32-wide k-step; B stream is
// amortized over two row-tiles. Batched via blockIdx.z (element strides).
// C = alpha * (A @ W^T) + bias ; written as f32 (Cf) and/or bf16 (Cb).
// ---------------------------------------------------------------------------
__global__ void __launch_bounds__(128)
gemm_bf16_nt(const bf16* __restrict__ A, const bf16* __restrict__ W,
             const float* __restrict__ bias,
             float* __restrict__ Cf, bf16* __restrict__ Cb,
             int M, int Nn, int K,
             long long sA, long long sW, long long sC, float alpha)
{
    const int wave = threadIdx.x >> 5;
    const int lane = threadIdx.x & 31;
    const int half = lane >> 4;
    const int r    = lane & 15;
    const int bz   = blockIdx.z;
    const int n0   = blockIdx.x * 64;
    const int m0   = blockIdx.y * 128 + wave * 32;

    const bf16* Ab = A + bz * sA;
    const bf16* Wb = W + bz * sW;

    int ar0 = m0 + r;       if (ar0 >= M) ar0 = M - 1;   // ragged M (S=100)
    int ar1 = m0 + 16 + r;  if (ar1 >= M) ar1 = M - 1;

    const bf16* pa0 = Ab + (long long)ar0 * K + half * 8;
    const bf16* pa1 = Ab + (long long)ar1 * K + half * 8;
    const bf16* pb0 = Wb + (long long)(n0 +      r) * K + half * 8;
    const bf16* pb1 = Wb + (long long)(n0 + 16 + r) * K + half * 8;
    const bf16* pb2 = Wb + (long long)(n0 + 32 + r) * K + half * 8;
    const bf16* pb3 = Wb + (long long)(n0 + 48 + r) * K + half * 8;

    v8f acc[8] = {};

    for (int k0 = 0; k0 < K; k0 += 32) {
        const v16bf a0 = ldfrag(pa0);
        const v16bf a1 = ldfrag(pa1);

        v16bf b = ldfrag(pb0);
        acc[0] = __builtin_amdgcn_wmma_f32_16x16x32_bf16(false, a0, false, b, (short)0, acc[0], false, false);
        acc[4] = __builtin_amdgcn_wmma_f32_16x16x32_bf16(false, a1, false, b, (short)0, acc[4], false, false);
        b = ldfrag(pb1);
        acc[1] = __builtin_amdgcn_wmma_f32_16x16x32_bf16(false, a0, false, b, (short)0, acc[1], false, false);
        acc[5] = __builtin_amdgcn_wmma_f32_16x16x32_bf16(false, a1, false, b, (short)0, acc[5], false, false);
        b = ldfrag(pb2);
        acc[2] = __builtin_amdgcn_wmma_f32_16x16x32_bf16(false, a0, false, b, (short)0, acc[2], false, false);
        acc[6] = __builtin_amdgcn_wmma_f32_16x16x32_bf16(false, a1, false, b, (short)0, acc[6], false, false);
        b = ldfrag(pb3);
        acc[3] = __builtin_amdgcn_wmma_f32_16x16x32_bf16(false, a0, false, b, (short)0, acc[3], false, false);
        acc[7] = __builtin_amdgcn_wmma_f32_16x16x32_bf16(false, a1, false, b, (short)0, acc[7], false, false);

        pa0 += 32; pa1 += 32;
        pb0 += 32; pb1 += 32; pb2 += 32; pb3 += 32;

        // stream-ahead hints; locality=3 -> near (WGP) scope prefetch
        __builtin_prefetch(pb0 + 32, 0, 3);
        __builtin_prefetch(pb1 + 32, 0, 3);
        __builtin_prefetch(pb2 + 32, 0, 3);
        __builtin_prefetch(pb3 + 32, 0, 3);
        __builtin_prefetch(pa0 + 32, 0, 3);
        __builtin_prefetch(pa1 + 32, 0, 3);
    }

    // C/D layout: lanes 0-15 reg i -> row base+i, lanes 16-31 -> row base+8+i
#pragma unroll
    for (int g = 0; g < 2; ++g) {
#pragma unroll
        for (int t = 0; t < 4; ++t) {
            const int col = n0 + t * 16 + r;
            const float bv = bias ? bias[col] : 0.0f;
            const v8f a = acc[g * 4 + t];
#pragma unroll
            for (int i = 0; i < 8; ++i) {
                const int row = m0 + g * 16 + half * 8 + i;
                if (row < M) {
                    const float v = a[i] * alpha + bv;
                    const long long offc = bz * sC + (long long)row * Nn + col;
                    if (Cf) Cf[offc] = v;
                    if (Cb) Cb[offc] = to_bf16(v);
                }
            }
        }
    }
}

// ---------------------------------------------------------------------------
// Elementwise / reduction kernels
// ---------------------------------------------------------------------------

// vectorized f32 -> bf16: 8 elements per thread (32B in, 16B out)
__global__ void cvt_f32_bf16_vec8(const float* __restrict__ src,
                                  bf16* __restrict__ dst, long long count8) {
    long long i = (long long)blockIdx.x * blockDim.x + threadIdx.x;
    if (i >= count8) return;
    const float4* s = reinterpret_cast<const float4*>(src) + 2 * i;
    const float4 a = s[0], c = s[1];
    union { v8bf v; uint4 u; } o;
    o.v[0] = to_bf16(a.x); o.v[1] = to_bf16(a.y);
    o.v[2] = to_bf16(a.z); o.v[3] = to_bf16(a.w);
    o.v[4] = to_bf16(c.x); o.v[5] = to_bf16(c.y);
    o.v[6] = to_bf16(c.z); o.v[7] = to_bf16(c.w);
    *reinterpret_cast<uint4*>(dst + 8 * i) = o.u;
}

// x[b,n,d] f32  ->  xT[b,d,n] bf16 (so updates GEMM is NT form)
__global__ void __launch_bounds__(256)
transpose_cvt_x(const float* __restrict__ x, bf16* __restrict__ xT) {
    __shared__ float tile[32][33];
    const int bz = blockIdx.z;
    const float* xb = x  + (long long)bz * N_ * D_;
    bf16*      xtb  = xT + (long long)bz * D_ * N_;
    const int n0 = blockIdx.x * 32, d0 = blockIdx.y * 32;
    const int tx = threadIdx.x;
    for (int i = threadIdx.y; i < 32; i += 8)
        tile[i][tx] = xb[(long long)(n0 + i) * D_ + d0 + tx];
    __syncthreads();
    // packed 32-bit stores: two bf16 (consecutive n) per lane
    if (tx < 16) {
        for (int i = threadIdx.y; i < 32; i += 8) {
            const unsigned lo = (unsigned)__builtin_bit_cast(unsigned short, to_bf16(tile[2 * tx][i]));
            const unsigned hi = (unsigned)__builtin_bit_cast(unsigned short, to_bf16(tile[2 * tx + 1][i]));
            *reinterpret_cast<unsigned*>(&xtb[(long long)(d0 + i) * N_ + n0 + 2 * tx]) =
                lo | (hi << 16);
        }
    }
}

// slots = mu + |sigma| * noise  (hash-based Box-Muller; JAX PRNG stream is
// not reproducible outside XLA — compile-only loop, distributional match)
__global__ void init_slots_kernel(const float* __restrict__ mu,
                                  const float* __restrict__ sigma,
                                  float* __restrict__ slots,
                                  bf16* __restrict__ slots_bf) {
    long long idx = (long long)blockIdx.x * blockDim.x + threadIdx.x;
    if (idx >= (long long)BS_ * D_) return;
    const int d = (int)(idx & (D_ - 1));
    unsigned x = (unsigned)idx * 2654435761u + 0x9E3779B9u;
    x ^= x >> 16; x *= 0x7feb352du; x ^= x >> 15; x *= 0x846ca68bu; x ^= x >> 16;
    unsigned y = x * 0x9E3779B1u + 12345u;
    y ^= y >> 16; y *= 0x7feb352du; y ^= y >> 15;
    float u1 = (float)((x >> 8) + 1u) * (1.0f / 16777216.0f);
    float u2 = (float)(y >> 8) * (1.0f / 16777216.0f);
    float nrm = sqrtf(-2.0f * __logf(u1)) * __cosf(6.28318530718f * u2);
    float v = mu[d] + fabsf(sigma[d]) * nrm;
    slots[idx] = v;
    slots_bf[idx] = to_bf16(v);
}

// one block per (b,s) row of 1024; softmax in place + bf16 copy for next GEMM
__global__ void __launch_bounds__(256)
softmax_rows(float* __restrict__ attn, bf16* __restrict__ attn_bf) {
    const long long base = (long long)blockIdx.x * N_;
    const int tid = threadIdx.x;
    __shared__ float sm[256];
    float v[4], m = -1e30f;
#pragma unroll
    for (int i = 0; i < 4; ++i) {
        v[i] = attn[base + tid + i * 256];
        m = fmaxf(m, v[i]);
    }
    sm[tid] = m; __syncthreads();
    for (int s = 128; s > 0; s >>= 1) {
        if (tid < s) sm[tid] = fmaxf(sm[tid], sm[tid + s]);
        __syncthreads();
    }
    m = sm[0]; __syncthreads();
    float s = 0.f;
#pragma unroll
    for (int i = 0; i < 4; ++i) { v[i] = __expf(v[i] - m); s += v[i]; }
    sm[tid] = s; __syncthreads();
    for (int k = 128; k > 0; k >>= 1) {
        if (tid < k) sm[tid] += sm[tid + k];
        __syncthreads();
    }
    const float inv = 1.0f / sm[0];
#pragma unroll
    for (int i = 0; i < 4; ++i) {
        const float o = v[i] * inv;
        attn   [base + tid + i * 256] = o;
        attn_bf[base + tid + i * 256] = to_bf16(o);
    }
}

// PyTorch GRU cell elementwise tail: slots <- (1-z)*n + z*slots
__global__ void gru_elementwise(const float* __restrict__ gi,
                                const float* __restrict__ gh,
                                float* __restrict__ slots,
                                bf16* __restrict__ slots_bf) {
    long long idx = (long long)blockIdx.x * blockDim.x + threadIdx.x;
    if (idx >= (long long)BS_ * D_) return;
    const long long row = idx >> 9;          // / D_
    const int d = (int)(idx & (D_ - 1));
    const long long g = row * TD_;
    const float ir = gi[g + d],          hr = gh[g + d];
    const float iz = gi[g + D_ + d],     hz = gh[g + D_ + d];
    const float in_ = gi[g + 2*D_ + d],  hn = gh[g + 2*D_ + d];
    const float r = 1.0f / (1.0f + __expf(-(ir + hr)));
    const float z = 1.0f / (1.0f + __expf(-(iz + hz)));
    const float nn = tanhf(in_ + r * hn);
    const float h = (1.0f - z) * nn + z * slots[idx];
    slots[idx] = h;
    slots_bf[idx] = to_bf16(h);
}

// out[b*S+s] = sum_d updates ; out[B*S] = slot_loss = sum(softmax)/(b*S*n) = 1/n
__global__ void __launch_bounds__(256)
rowsum_out(const float* __restrict__ updates, float* __restrict__ out) {
    const long long base = (long long)blockIdx.x * D_;
    const int tid = threadIdx.x;
    __shared__ float sm[256];
    sm[tid] = updates[base + tid] + updates[base + tid + 256];
    __syncthreads();
    for (int k = 128; k > 0; k >>= 1) {
        if (tid < k) sm[tid] += sm[tid + k];
        __syncthreads();
    }
    if (tid == 0) out[blockIdx.x] = sm[0];
    if (blockIdx.x == 0 && tid == 0) out[BS_] = 1.0f / (float)N_;
}

// ---------------------------------------------------------------------------
extern "C" void kernel_launch(void* const* d_in, const int* in_sizes, int n_in,
                              void* d_out, int out_size, void* d_ws, size_t ws_size,
                              hipStream_t stream) {
    const float* inputs   = (const float*)d_in[0];
    const float* inputs_x = (const float*)d_in[1];
    const float* slots_mu = (const float*)d_in[2];
    const float* slots_sg = (const float*)d_in[3];
    const float* to_q_w   = (const float*)d_in[4];
    const float* to_q_b   = (const float*)d_in[5];
    const float* to_k_w   = (const float*)d_in[6];
    const float* to_k_b   = (const float*)d_in[7];
    const float* w_ih     = (const float*)d_in[8];
    const float* w_hh     = (const float*)d_in[9];
    const float* b_ih     = (const float*)d_in[10];
    const float* b_hh     = (const float*)d_in[11];
    float* out = (float*)d_out;

    // ---- workspace carve-out (bytes, 256-aligned) ----
    char*  ws  = (char*)d_ws;
    size_t off = 0;
    auto carve = [&](size_t bytes) -> void* {
        void* p = ws + off;
        off = (off + bytes + 255) & ~(size_t)255;
        return p;
    };
    bf16*  ibf      = (bf16*) carve((size_t)B_ * N_ * D_ * 2);  // inputs bf16
    bf16*  xT       = (bf16*) carve((size_t)B_ * D_ * N_ * 2);  // inputs_x^T bf16
    bf16*  kbf      = (bf16*) carve((size_t)B_ * N_ * D_ * 2);  // k bf16
    bf16*  wqbf     = (bf16*) carve((size_t)D_ * D_ * 2);
    bf16*  wkbf     = (bf16*) carve((size_t)D_ * D_ * 2);
    bf16*  wihbf    = (bf16*) carve((size_t)TD_ * D_ * 2);
    bf16*  whhbf    = (bf16*) carve((size_t)TD_ * D_ * 2);
    float* slots    = (float*)carve((size_t)BS_ * D_ * 4);
    bf16*  slots_bf = (bf16*) carve((size_t)BS_ * D_ * 2);
    bf16*  qbf      = (bf16*) carve((size_t)BS_ * D_ * 2);
    float* attn     = (float*)carve((size_t)BS_ * N_ * 4);
    bf16*  attn_bf  = (bf16*) carve((size_t)BS_ * N_ * 2);
    float* updates  = (float*)carve((size_t)BS_ * D_ * 4);
    bf16*  upd_bf   = (bf16*) carve((size_t)BS_ * D_ * 2);
    float* gi       = (float*)carve((size_t)BS_ * TD_ * 4);
    float* gh       = (float*)carve((size_t)BS_ * TD_ * 4);
    (void)ws_size; (void)in_sizes; (void)n_in; (void)out_size;

    // ---- precompute: bf16 inputs/weights, transposed x, initial slots, k ----
    {
        const long long c8 = (long long)B_ * N_ * D_ / 8;
        cvt_f32_bf16_vec8<<<(unsigned)(c8 / 256), 256, 0, stream>>>(inputs, ibf, c8);
    }
    cvt_f32_bf16_vec8<<<(D_*D_/8)/256, 256, 0, stream>>>(to_q_w, wqbf, D_*D_/8);
    cvt_f32_bf16_vec8<<<(D_*D_/8)/256, 256, 0, stream>>>(to_k_w, wkbf, D_*D_/8);
    cvt_f32_bf16_vec8<<<(TD_*D_/8)/256, 256, 0, stream>>>(w_ih, wihbf, TD_*D_/8);
    cvt_f32_bf16_vec8<<<(TD_*D_/8)/256, 256, 0, stream>>>(w_hh, whhbf, TD_*D_/8);

    transpose_cvt_x<<<dim3(N_/32, D_/32, B_), dim3(32, 8), 0, stream>>>(inputs_x, xT);

    init_slots_kernel<<<(BS_*D_)/256, 256, 0, stream>>>(slots_mu, slots_sg,
                                                        slots, slots_bf);

    // k = inputs @ Wk^T + bk   (bf16 out)
    gemm_bf16_nt<<<dim3(D_/64, (B_*N_)/128, 1), 128, 0, stream>>>(
        ibf, wkbf, to_k_b, nullptr, kbf,
        B_*N_, D_, D_, 0, 0, 0, 1.0f);

    // ---- 3 slot-attention iterations ----
    for (int it = 0; it < 3; ++it) {
        // q = slots @ Wq^T + bq  (bf16 out)
        gemm_bf16_nt<<<dim3(D_/64, BS_/128, 1), 128, 0, stream>>>(
            slots_bf, wqbf, to_q_b, nullptr, qbf,
            BS_, D_, D_, 0, 0, 0, 1.0f);

        // dots = scale * q @ k^T   (batched over b; M=100 is ragged)
        gemm_bf16_nt<<<dim3(N_/64, 1, B_), 128, 0, stream>>>(
            qbf, kbf, nullptr, attn, nullptr,
            S_, N_, D_,
            (long long)S_*D_, (long long)N_*D_, (long long)S_*N_, SCALE_);

        softmax_rows<<<BS_, 256, 0, stream>>>(attn, attn_bf);

        // updates = (attn @ x) / d   via xT:  attn[S,N] @ xT[D,N]^T
        gemm_bf16_nt<<<dim3(D_/64, 1, B_), 128, 0, stream>>>(
            attn_bf, xT, nullptr, updates, upd_bf,
            S_, D_, N_,
            (long long)S_*N_, (long long)D_*N_, (long long)S_*D_,
            1.0f / (float)D_);

        // GRU gate GEMMs
        gemm_bf16_nt<<<dim3(TD_/64, BS_/128, 1), 128, 0, stream>>>(
            upd_bf, wihbf, b_ih, gi, nullptr,
            BS_, TD_, D_, 0, 0, 0, 1.0f);
        gemm_bf16_nt<<<dim3(TD_/64, BS_/128, 1), 128, 0, stream>>>(
            slots_bf, whhbf, b_hh, gh, nullptr,
            BS_, TD_, D_, 0, 0, 0, 1.0f);

        gru_elementwise<<<(BS_*D_)/256, 256, 0, stream>>>(gi, gh, slots, slots_bf);
    }

    // out[b,s] = sum_d updates ; out[B*S] = 1/n (softmax rows sum to 1)
    rowsum_out<<<BS_, 256, 0, stream>>>(updates, out);
}
